// RecursiveEncoder_80951543595071
// MI455X (gfx1250) — matrix operand order repaired and checked by reading, not compiled, not run on hardware
//
#include <hip/hip_runtime.h>
#include <hip/hip_bf16.h>

typedef __attribute__((ext_vector_type(16))) _Float16 v16h;
typedef __attribute__((ext_vector_type(8)))  float    v8f;

#define B_SZ   16
#define H_SZ   256
#define N_LEAF 15625
#define N_INT  3906

union AU { uint4 q[2]; v16h h; };

// ---------------------------------------------------------------------------
// 1) Transpose+convert W_node (3,1280,256) f32 -> Wt (3,256,1280) f16
// ---------------------------------------------------------------------------
__global__ __launch_bounds__(256) void wnode_prep_kernel(
    const float* __restrict__ Wn, _Float16* __restrict__ Wt) {
  int idx = blockIdx.x * 256 + threadIdx.x;           // 3*256*1280 = 983040
  if (idx >= 3 * 256 * 1280) return;
  int t   = idx / (256 * 1280);
  int rem = idx - t * (256 * 1280);
  int col = rem / 1280;
  int k   = rem - col * 1280;
  Wt[idx] = (_Float16)Wn[(size_t)(t * 1280 + k) * 256 + col];
}

// ---------------------------------------------------------------------------
// 2) Leaf encoder: feats = relu(box@W_box+b) + relu(sem@W_sem+b), stored f16
// ---------------------------------------------------------------------------
__global__ __launch_bounds__(256) void leaf_kernel(
    const float* __restrict__ lbox, const float* __restrict__ lsem,
    const float* __restrict__ Wbox, const float* __restrict__ bbox,
    const float* __restrict__ Wsem, const float* __restrict__ bsem,
    _Float16* __restrict__ out) {
  const int h = threadIdx.x;
  const int row0 = blockIdx.x * 125;
  float wb[4], ws[16];
#pragma unroll
  for (int k = 0; k < 4; ++k)  wb[k] = Wbox[k * H_SZ + h];
#pragma unroll
  for (int k = 0; k < 16; ++k) ws[k] = Wsem[k * H_SZ + h];
  const float bb = bbox[h], bs = bsem[h];
  __shared__ float sb[20];
  for (int r = 0; r < 125; ++r) {
    const int g = row0 + r;
    __syncthreads();
    if (h < 4)        sb[h] = lbox[(size_t)g * 4 + h];
    else if (h < 20)  sb[h] = lsem[(size_t)g * 16 + (h - 4)];
    __syncthreads();
    float vb = bb;
#pragma unroll
    for (int k = 0; k < 4; ++k)  vb += sb[k] * wb[k];
    float vs = bs;
#pragma unroll
    for (int k = 0; k < 16; ++k) vs += sb[4 + k] * ws[k];
    vb = fmaxf(vb, 0.f);
    vs = fmaxf(vs, 0.f);
    out[(size_t)g * H_SZ + h] = (_Float16)(vb + vs);
  }
}

// ---------------------------------------------------------------------------
// 3) Tree level. Block = 32-row x 256-col tile (two 16-row WMMA tiles sharing
//    every B operand). 8 waves; wave owns 2 col-tiles x 2 row-tiles = 4 accs.
//    Per-row weight selection via masked-A accumulation: C = sum_t A_t @ W_t.
// ---------------------------------------------------------------------------
__global__ __launch_bounds__(256) void level_kernel(
    const _Float16* __restrict__ A,      // (rows, 1280) f16, contiguous
    _Float16*       __restrict__ Y,      // (rows, 256)  f16
    const _Float16* __restrict__ Wt,     // (3,256,1280) f16, col-major-K
    const float*    __restrict__ bnode,  // (3,256)
    const int*      __restrict__ ntype,  // (B, N_INT)
    const float*    __restrict__ ibox,   // (B, N_INT, 4)
    const float*    __restrict__ Wbox,   // (4,256)
    const float*    __restrict__ bbox,   // (256)
    int n, int off, int rows) {
  const int tid    = threadIdx.x;
  const int lane   = tid & 31;
  const int wave   = tid >> 5;
  const int mrow   = lane & 15;
  const int halfid = lane >> 4;
  const int rt     = blockIdx.x * 32;
  const int col0   = wave * 32 + mrow;
  const int col1   = col0 + 16;

  // Row indices for this lane's two A-rows (tile1 clamped; stores predicated)
  const int g0 = rt + mrow;
  const int g1raw = rt + 16 + mrow;
  const int g1 = (g1raw < rows) ? g1raw : (rows - 1);

  const _Float16* pA0 = A + (size_t)g0 * 1280 + halfid * 8;
  const _Float16* pA1 = A + (size_t)g1 * 1280 + halfid * 8;

  const _Float16* pB[3][2];
#pragma unroll
  for (int t = 0; t < 3; ++t) {
    pB[t][0] = Wt + ((size_t)(t * 256 + col0)) * 1280 + halfid * 8;
    pB[t][1] = Wt + ((size_t)(t * 256 + col1)) * 1280 + halfid * 8;
  }

  // node types for the two A rows (mask uniform over K)
  int bA = g0 / n;
  const int myt0 = ntype[bA * N_INT + off + (g0 - bA * n)];
  bA = g1 / n;
  const int myt1 = ntype[bA * N_INT + off + (g1 - bA * n)];

  v8f  c00 = {};
  v8f  c01 = {};
  v8f  c10 = {};
  v8f  c11 = {};
  v16h z   = {};

  for (int k = 0; k < 1280; k += 32) {
    __builtin_prefetch(pA0 + k + 128, 0, 3);
    __builtin_prefetch(pA1 + k + 128, 0, 3);
    AU a0, a1;
    a0.q[0] = *(const uint4*)(pA0 + k);
    a0.q[1] = *(const uint4*)(pA0 + k + 16);
    a1.q[0] = *(const uint4*)(pA1 + k);
    a1.q[1] = *(const uint4*)(pA1 + k + 16);
#pragma unroll
    for (int t = 0; t < 3; ++t) {
      const v16h at0 = (myt0 == t) ? a0.h : z;
      const v16h at1 = (myt1 == t) ? a1.h : z;
      AU bu0, bu1;
      bu0.q[0] = *(const uint4*)(pB[t][0] + k);
      bu0.q[1] = *(const uint4*)(pB[t][0] + k + 16);
      bu1.q[0] = *(const uint4*)(pB[t][1] + k);
      bu1.q[1] = *(const uint4*)(pB[t][1] + k + 16);
      c00 = __builtin_amdgcn_wmma_f32_16x16x32_f16(false, at0, false, bu0.h,
                                                   (short)0, c00, false, false);
      c10 = __builtin_amdgcn_wmma_f32_16x16x32_f16(false, at1, false, bu0.h,
                                                   (short)0, c10, false, false);
      c01 = __builtin_amdgcn_wmma_f32_16x16x32_f16(false, at0, false, bu1.h,
                                                   (short)0, c01, false, false);
      c11 = __builtin_amdgcn_wmma_f32_16x16x32_f16(false, at1, false, bu1.h,
                                                   (short)0, c11, false, false);
    }
  }

  // Epilogue: bias(select by type) + relu + internal box encoder, store f16
  const float w00 = Wbox[col0], w10 = Wbox[256 + col0],
              w20 = Wbox[512 + col0], w30 = Wbox[768 + col0], bc0 = bbox[col0];
  const float w01 = Wbox[col1], w11 = Wbox[256 + col1],
              w21 = Wbox[512 + col1], w31 = Wbox[768 + col1], bc1 = bbox[col1];
#pragma unroll
  for (int tile = 0; tile < 2; ++tile) {
#pragma unroll
    for (int r = 0; r < 8; ++r) {
      const int m  = r + halfid * 8;
      const int g  = rt + tile * 16 + m;
      if (g >= rows) continue;
      const int bi = g / n;
      const int j  = g - bi * n;
      const int ni = bi * N_INT + off + j;
      const int t  = ntype[ni];
      const float* ib = ibox + (size_t)ni * 4;
      const float i0 = ib[0], i1 = ib[1], i2 = ib[2], i3 = ib[3];
      const float box0 = fmaxf(i0*w00 + i1*w10 + i2*w20 + i3*w30 + bc0, 0.f);
      const float box1 = fmaxf(i0*w01 + i1*w11 + i2*w21 + i3*w31 + bc1, 0.f);
      const float v0 = (tile == 0) ? c00[r] : c10[r];
      const float v1 = (tile == 0) ? c01[r] : c11[r];
      const float y0 = fmaxf(v0 + bnode[t * 256 + col0], 0.f);
      const float y1 = fmaxf(v1 + bnode[t * 256 + col1], 0.f);
      Y[(size_t)g * H_SZ + col0] = (_Float16)(y0 + box0);
      Y[(size_t)g * H_SZ + col1] = (_Float16)(y1 + box1);
    }
  }
}

// ---------------------------------------------------------------------------
// 4) VAE head: enc=relu(root@W1+b1); mu/logvar; out=[eps*std+mu, kld]
// ---------------------------------------------------------------------------
__global__ __launch_bounds__(256) void head_kernel(
    const _Float16* __restrict__ root, const float* __restrict__ W1,
    const float* __restrict__ b1v, const float* __restrict__ Wmu,
    const float* __restrict__ bmuv, const float* __restrict__ Wvar,
    const float* __restrict__ bvarv, const float* __restrict__ eps,
    float* __restrict__ out) {
  __shared__ float sr[256];
  __shared__ float se[256];
  const int b = blockIdx.x;
  const int h = threadIdx.x;
  sr[h] = (float)root[(size_t)b * H_SZ + h];
  __syncthreads();
  float acc = b1v[h];
  for (int k = 0; k < 256; ++k) acc += sr[k] * W1[k * 256 + h];
  se[h] = fmaxf(acc, 0.f);
  __syncthreads();
  float mu = bmuv[h], lv = bvarv[h];
  for (int k = 0; k < 256; ++k) {
    const float e = se[k];
    mu += e * Wmu[k * 256 + h];
    lv += e * Wvar[k * 256 + h];
  }
  const float stdv = expf(0.5f * lv);
  out[(size_t)b * 512 + h]       = eps[(size_t)b * 256 + h] * stdv + mu;
  out[(size_t)b * 512 + 256 + h] = 1.0f + lv - mu * mu - expf(lv);
}

// ---------------------------------------------------------------------------
extern "C" void kernel_launch(void* const* d_in, const int* in_sizes, int n_in,
                              void* d_out, int out_size, void* d_ws, size_t ws_size,
                              hipStream_t stream) {
  (void)in_sizes; (void)n_in; (void)out_size; (void)ws_size;
  const float* leaf_box = (const float*)d_in[0];
  const float* leaf_sem = (const float*)d_in[1];
  const float* int_box  = (const float*)d_in[2];
  const int*   ntype    = (const int*)  d_in[3];
  const float* eps      = (const float*)d_in[4];
  const float* W_box    = (const float*)d_in[5];
  const float* b_box    = (const float*)d_in[6];
  const float* W_sem    = (const float*)d_in[7];
  const float* b_sem    = (const float*)d_in[8];
  const float* W_node   = (const float*)d_in[9];
  const float* b_node   = (const float*)d_in[10];
  const float* W1       = (const float*)d_in[11];
  const float* b1v      = (const float*)d_in[12];
  const float* Wmu      = (const float*)d_in[13];
  const float* bmu      = (const float*)d_in[14];
  const float* Wvar     = (const float*)d_in[15];
  const float* bvar     = (const float*)d_in[16];
  float* out = (float*)d_out;

  // workspace layout
  char* ws = (char*)d_ws;
  _Float16* wt   = (_Float16*)(ws);                                  // 2 MB
  _Float16* buf0 = (_Float16*)(ws + (size_t)(2u << 20));             // 128 MB
  _Float16* buf1 = (_Float16*)(ws + (size_t)(2u << 20) + 128000000ull); // 25.6 MB

  // 1) weight prep (3*256*1280 = 983040 elems)
  wnode_prep_kernel<<<3840, 256, 0, stream>>>(W_node, wt);

  // 2) leaf features -> buf0 (level "6" parity: even -> buf0)
  leaf_kernel<<<(B_SZ * N_LEAF) / 125, 256, 0, stream>>>(
      leaf_box, leaf_sem, W_box, b_box, W_sem, b_sem, buf0);

  // 3) contract the tree, levels 5..0 (ping-pong: level l -> (l odd ? buf1 : buf0))
  const int npow[6] = {1, 5, 25, 125, 625, 3125};
  const int offs[6] = {0, 1, 6, 31, 156, 781};
  for (int l = 5; l >= 0; --l) {
    const _Float16* child = (((l + 1) & 1) ? buf1 : buf0);
    _Float16*       outb  = ((l & 1) ? buf1 : buf0);
    const int rows = B_SZ * npow[l];
    const int grid = (rows + 31) / 32;
    level_kernel<<<grid, 256, 0, stream>>>(
        child, outb, wt, b_node, ntype, int_box, W_box, b_box,
        npow[l], offs[l], rows);
  }

  // 4) head: root lives in buf0 (level 0, rows 0..15)
  head_kernel<<<B_SZ, 256, 0, stream>>>(buf0, W1, b1v, Wmu, bmu, Wvar, bvar,
                                        eps, out);
}
